// HybridAttention_24644522344905
// MI455X (gfx1250) — matrix operand order repaired and checked by reading, not compiled, not run on hardware
//
#include <hip/hip_runtime.h>

typedef __bf16 bf16_t;
typedef __attribute__((ext_vector_type(16))) __bf16 v16bf;
typedef __attribute__((ext_vector_type(8)))  float   v8f;
typedef __attribute__((ext_vector_type(4)))  int     v4i_t;

union BfFrag { uint4 u[2]; v16bf v; };

#if defined(__has_builtin)
#if __has_builtin(__builtin_amdgcn_global_load_async_to_lds_b128)
#define USE_ASYNC_COPY 1
#endif
#endif

#ifdef USE_ASYNC_COPY
__device__ __forceinline__ void async_b128(const void* g, void* l) {
  __builtin_amdgcn_global_load_async_to_lds_b128(
      (__attribute__((address_space(1))) v4i_t*)g,
      (__attribute__((address_space(3))) v4i_t*)l, 0, 0);
}
__device__ __forceinline__ void wait_async() {
#if __has_builtin(__builtin_amdgcn_s_wait_asynccnt)
  __builtin_amdgcn_s_wait_asynccnt(0);
#else
  asm volatile("s_wait_asynccnt 0x0" ::: "memory");
#endif
}
#endif

// ---------------- conversion / packing kernels ----------------

__global__ void k_cvt_bf16(const float* __restrict__ src, bf16_t* __restrict__ dst, int n4) {
  int i = blockIdx.x * blockDim.x + threadIdx.x;
  if (i >= n4) return;
  float4 v = reinterpret_cast<const float4*>(src)[i];
  bf16_t* d = dst + (size_t)i * 4;
  d[0] = (bf16_t)v.x; d[1] = (bf16_t)v.y; d[2] = (bf16_t)v.z; d[3] = (bf16_t)v.w;
}

// out[k*256 + n] = qw[n*256 + k]   (B operand for q = x @ qw^T)
__global__ void k_pack_qw(const float* __restrict__ qw, bf16_t* __restrict__ out) {
  int idx = blockIdx.x * 256 + threadIdx.x;
  int n = idx & 255, k = idx >> 8;
  out[idx] = (bf16_t)qw[n * 256 + k];
}

// out[((ky*KH+kx)*256 + c)*512 + o] = cw[((o*256 + c)*KH + ky)*KH + kx]
__global__ void k_pack_conv(const float* __restrict__ cw, bf16_t* __restrict__ out, int KH) {
  int idx = blockIdx.x * 256 + threadIdx.x;
  int o = idx & 511; int t = idx >> 9; int c = t & 255; int t2 = t >> 8;
  int kx = t2 % KH; int ky = t2 / KH;
  out[idx] = (bf16_t)cw[((o * 256 + c) * KH + ky) * KH + kx];
}

// out[k*256 + n] = pw[n*512 + k]
__global__ void k_pack_pw(const float* __restrict__ pw, bf16_t* __restrict__ out) {
  int idx = blockIdx.x * 256 + threadIdx.x; // 512*256
  int n = idx & 255, k = idx >> 8;
  out[idx] = (bf16_t)pw[n * 512 + k];
}

// ---------------- tiled bf16 WMMA GEMM ----------------
// MODE 0: A row-major [M][lda]
// MODE 1: level-0 conv gather: m=(b,py,px) in 8x16x16, k=(ky,kx,c) in 8x8x256, pixel stride lda
// MODE 2: level-1 conv gather: m=(b,py,px) in 8x32x32, k=(ky,kx,c) in 4x4x256, pixel stride lda
template<int MODE>
__device__ __forceinline__ const bf16_t* aAddr(const bf16_t* A, int m, int k, int lda) {
  if constexpr (MODE == 0) {
    return A + (size_t)m * lda + k;
  } else if constexpr (MODE == 1) {
    int b = m >> 8, py = (m >> 4) & 15, px = m & 15;
    int ky = k >> 11, kx = (k >> 8) & 7, c = k & 255;
    return A + (size_t)((b * 128 + py * 8 + ky) * 128 + px * 8 + kx) * lda + c;
  } else {
    int b = m >> 10, py = (m >> 5) & 31, px = m & 31;
    int ky = k >> 10, kx = (k >> 8) & 3, c = k & 255;
    return A + (size_t)((b * 128 + py * 4 + ky) * 128 + px * 4 + kx) * lda + c;
  }
}

template<int MODE>
__global__ __launch_bounds__(256) void k_gemm(const bf16_t* __restrict__ A,
                                              const bf16_t* __restrict__ B,
                                              float* __restrict__ C,
                                              const float* __restrict__ bias,
                                              int K, int lda, int ldb, int ldc) {
  // A tile rows: 32 K-elems = 64B data, padded to 80B (16B aligned, near conflict-free)
  // B tile rows: 128 N-elems = 256B data, padded to 272B
  __shared__ __align__(16) unsigned char sA[2][128 * 80];
  __shared__ __align__(16) unsigned char sB[2][32 * 272];

  const int tid   = threadIdx.x;
  const int lane  = tid & 31;
  const int wave  = tid >> 5;
  const int waveM = wave & 3;        // 4 waves along M (32 rows each)
  const int waveN = wave >> 2;       // 2 waves along N (64 cols each)
  const int mBase = blockIdx.x * 128;
  const int nBase = blockIdx.y * 128;

  const int arow = tid >> 1, ahalf = tid & 1;   // A stager: 128 rows x 2 halves
  const int brow = tid >> 3, bseg  = tid & 7;   // B stager: 32 rows x 8 segs

#ifdef USE_ASYNC_COPY
  // Async global->LDS staging (ASYNCcnt-tracked), no VGPR bounce, one barrier/step.
  auto issueTile = [&](int k0, int buf) {
    const bf16_t* ap = aAddr<MODE>(A, mBase + arow, k0 + ahalf * 16, lda);
    async_b128(ap,     &sA[buf][arow * 80 + ahalf * 32]);
    async_b128(ap + 8, &sA[buf][arow * 80 + ahalf * 32 + 16]);
    const bf16_t* bp = B + (size_t)(k0 + brow) * ldb + nBase + bseg * 16;
    async_b128(bp,     &sB[buf][brow * 272 + bseg * 32]);
    async_b128(bp + 8, &sB[buf][brow * 272 + bseg * 32 + 16]);
  };
#else
  uint4 ar0, ar1, br0, br1;
  auto loadG = [&](int k0) {
    const bf16_t* ap = aAddr<MODE>(A, mBase + arow, k0 + ahalf * 16, lda);
    ar0 = *reinterpret_cast<const uint4*>(ap);
    ar1 = *reinterpret_cast<const uint4*>(ap + 8);
    const bf16_t* bp = B + (size_t)(k0 + brow) * ldb + nBase + bseg * 16;
    br0 = *reinterpret_cast<const uint4*>(bp);
    br1 = *reinterpret_cast<const uint4*>(bp + 8);
    __builtin_prefetch(bp + (size_t)32 * ldb, 0, 1);   // next K-step (speculative)
  };
  auto storeS = [&](int buf) {
    *reinterpret_cast<uint4*>(&sA[buf][arow * 80 + ahalf * 32])      = ar0;
    *reinterpret_cast<uint4*>(&sA[buf][arow * 80 + ahalf * 32 + 16]) = ar1;
    *reinterpret_cast<uint4*>(&sB[buf][brow * 272 + bseg * 32])      = br0;
    *reinterpret_cast<uint4*>(&sB[buf][brow * 272 + bseg * 32 + 16]) = br1;
  };
#endif

  v8f acc[2][4];
  #pragma unroll
  for (int i = 0; i < 2; i++)
    #pragma unroll
    for (int j = 0; j < 4; j++)
      #pragma unroll
      for (int e = 0; e < 8; e++) acc[i][j][e] = 0.f;

  const int nk = K >> 5;
  const int laneR  = lane & 15;
  const int laneHi = lane & 16;   // byte offset 0/16 selecting K-half per ISA A layout

#ifdef USE_ASYNC_COPY
  issueTile(0, 0);
  wait_async();
  __syncthreads();
#else
  loadG(0);
  storeS(0);
  __syncthreads();
#endif

  for (int s = 0; s < nk; ++s) {
#ifdef USE_ASYNC_COPY
    if (s + 1 < nk) issueTile((s + 1) << 5, (s + 1) & 1);
#else
    if (s + 1 < nk) loadG((s + 1) << 5);
#endif
    const unsigned char* As = sA[s & 1];
    const unsigned char* Bs = sB[s & 1];
    BfFrag af[2], bfg[4];
    #pragma unroll
    for (int i = 0; i < 2; i++) {
      int base = (waveM * 32 + i * 16 + laneR) * 80 + laneHi;
      af[i].u[0] = *reinterpret_cast<const uint4*>(As + base);       // K 0..7 / 8..15
      af[i].u[1] = *reinterpret_cast<const uint4*>(As + base + 32);  // K 16..23 / 24..31
    }
    #pragma unroll
    for (int j = 0; j < 4; j++) {
      int base = (lane >> 4) * 8 * 272 + (waveN * 64 + j * 16 + laneR) * 2;   // n = laneR, k-row 8h+...
      #pragma unroll
      for (int e = 0; e < 16; e++)
        bfg[j].v[e] = *reinterpret_cast<const bf16_t*>(Bs + base + ((e & 7) + (e >> 3) * 16) * 272);
    }
    #pragma unroll
    for (int i = 0; i < 2; i++)
      #pragma unroll
      for (int j = 0; j < 4; j++)
        acc[i][j] = __builtin_amdgcn_wmma_f32_16x16x32_bf16(
            false, af[i].v, false, bfg[j].v, (short)0, acc[i][j], false, false);
#ifdef USE_ASYNC_COPY
    wait_async();        // our copies into the next buffer have landed
    __syncthreads();     // everyone done reading current buffer; copies visible
#else
    __syncthreads();
    if (s + 1 < nk) { storeS((s + 1) & 1); __syncthreads(); }
#endif
  }

  const int rAdd = (lane & 16) ? 8 : 0;   // C/D layout: lanes 16-31 hold M=r+8
  #pragma unroll
  for (int i = 0; i < 2; i++) {
    #pragma unroll
    for (int j = 0; j < 4; j++) {
      int col  = nBase + waveN * 64 + j * 16 + laneR;
      int row0 = mBase + waveM * 32 + i * 16 + rAdd;
      float bv = bias ? bias[col] : 0.f;
      #pragma unroll
      for (int r = 0; r < 8; r++)
        C[(size_t)(row0 + r) * ldc + col] = acc[i][j][r] + bv;
    }
  }
}

// ---------------- bias + LayerNorm (in place, 512 channels, wave per row) --------

__global__ __launch_bounds__(256) void k_bias_ln(float* __restrict__ data,
                                                 const float* __restrict__ cb,
                                                 const float* __restrict__ g,
                                                 const float* __restrict__ bta) {
  const int lane = threadIdx.x & 31;
  const int wave = threadIdx.x >> 5;
  const int row  = blockIdx.x * 8 + wave;
  float* rp = data + (size_t)row * 512;
  float v[16];
  float s = 0.f, s2 = 0.f;
  #pragma unroll
  for (int u = 0; u < 4; ++u) {
    float4 t = reinterpret_cast<float4*>(rp)[lane * 4 + u];
    int c = lane * 16 + u * 4;
    t.x += cb[c]; t.y += cb[c + 1]; t.z += cb[c + 2]; t.w += cb[c + 3];
    v[u * 4 + 0] = t.x; v[u * 4 + 1] = t.y; v[u * 4 + 2] = t.z; v[u * 4 + 3] = t.w;
    s  += t.x + t.y + t.z + t.w;
    s2 += t.x * t.x + t.y * t.y + t.z * t.z + t.w * t.w;
  }
  #pragma unroll
  for (int m = 16; m >= 1; m >>= 1) { s += __shfl_xor(s, m, 32); s2 += __shfl_xor(s2, m, 32); }
  float mean = s * (1.f / 512.f);
  float var  = s2 * (1.f / 512.f) - mean * mean;    // biased var, matches torch LN
  float rstd = rsqrtf(var + 1e-5f);
  #pragma unroll
  for (int u = 0; u < 16; ++u) {
    int c = lane * 16 + (u >> 2) * 4 + (u & 3);
    rp[c] = (v[u] - mean) * rstd * g[c] + bta[c];
  }
}

// ---------------- windowed attention (4 KV tokens per window, VALU path) ---------

template<int LEVEL>
__global__ void k_attn(const float* __restrict__ qA, const float* __restrict__ kv,
                       bf16_t* __restrict__ yb) {
  constexpr int WSZ   = (LEVEL == 0) ? 16 : 8;
  constexpr int TOK   = WSZ * WSZ;       // 256 or 64 (== blockDim.x)
  constexpr int chOff = (LEVEL == 0) ? 0 : 256;
  constexpr int KVG   = (LEVEL == 0) ? 16 : 32;   // conv output grid per side
  __shared__ float sKV[2][4][32];        // [K/V][token][dim]

  const int bi = blockIdx.x;
  int ww, wh, h, b;
  if (LEVEL == 0) { ww = bi & 7;  wh = (bi >> 3) & 7;  h = (bi >> 6) & 7; b = bi >> 9;  }
  else            { ww = bi & 15; wh = (bi >> 4) & 15; h = (bi >> 8) & 7; b = bi >> 11; }

  const int t = threadIdx.x;
  for (int idx = t; idx < 256; idx += TOK) {
    int s = idx >> 6; int r = idx & 63; int sel = r >> 5; int d = r & 31;
    int py = 2 * wh + (s >> 1), px = 2 * ww + (s & 1);
    const float* rp = kv + (size_t)((b * KVG + py) * KVG + px) * 512;
    sKV[sel][s][d] = rp[sel * 256 + h * 32 + d];
  }
  __syncthreads();

  const float* qp;
  int orow;
  if (LEVEL == 0) {
    int ty = t >> 4, tx = t & 15;
    int y = wh * 16 + ty, x = ww * 16 + tx;
    orow = (b * 128 + y) * 128 + x;
    qp = qA + (size_t)orow * 256 + h * 32;
  } else {
    // bug-faithful flat remap of the level-0 permuted q view
    int ty = t >> 3, tx = t & 7;
    int y = wh * 8 + ty, x = ww * 8 + tx;
    orow = (b * 128 + y) * 128 + x;
    int gidx = (((wh * 8 + ty) * 16 + ww) * 8 + tx) * 8 + h;     // f >> 5
    int t0 = gidx & 255, ww0 = (gidx >> 8) & 7, wh0 = (gidx >> 11) & 7, h0 = (gidx >> 14) & 7;
    int y0 = wh0 * 16 + (t0 >> 4), x0 = ww0 * 16 + (t0 & 15);
    qp = qA + (size_t)((b * 128 + y0) * 128 + x0) * 256 + h0 * 32;
  }

  float q[32];
  #pragma unroll
  for (int u = 0; u < 8; u++) {
    float4 v4 = reinterpret_cast<const float4*>(qp)[u];
    q[u * 4] = v4.x; q[u * 4 + 1] = v4.y; q[u * 4 + 2] = v4.z; q[u * 4 + 3] = v4.w;
  }

  const float scale = 0.17677669529663687f;  // 32^-0.5
  float l[4];
  #pragma unroll
  for (int s = 0; s < 4; s++) {
    float a = 0.f;
    #pragma unroll
    for (int d = 0; d < 32; d++) a += q[d] * sKV[0][s][d];
    l[s] = a * scale;
  }
  float mx = fmaxf(fmaxf(l[0], l[1]), fmaxf(l[2], l[3]));
  float p[4], sum = 0.f;
  #pragma unroll
  for (int s = 0; s < 4; s++) { p[s] = __expf(l[s] - mx); sum += p[s]; }
  float inv = 1.f / sum;

  bf16_t* op = yb + (size_t)orow * 512 + chOff + h * 32;
  #pragma unroll
  for (int d = 0; d < 32; d++) {
    float o = (p[0] * sKV[1][0][d] + p[1] * sKV[1][1][d] +
               p[2] * sKV[1][2][d] + p[3] * sKV[1][3][d]) * inv;
    op[d] = (bf16_t)o;
  }
}

// ---------------- host side ----------------

extern "C" void kernel_launch(void* const* d_in, const int* in_sizes, int n_in,
                              void* d_out, int out_size, void* d_ws, size_t ws_size,
                              hipStream_t stream) {
  (void)in_sizes; (void)n_in; (void)out_size; (void)ws_size;
  const float* x   = (const float*)d_in[0];
  const float* qw  = (const float*)d_in[1];
  const float* cw0 = (const float*)d_in[2];
  const float* cb0 = (const float*)d_in[3];
  const float* g0  = (const float*)d_in[4];
  const float* b0  = (const float*)d_in[5];
  const float* cw1 = (const float*)d_in[6];
  const float* cb1 = (const float*)d_in[7];
  const float* g1  = (const float*)d_in[8];
  const float* b1  = (const float*)d_in[9];
  const float* pw  = (const float*)d_in[10];
  const float* pb  = (const float*)d_in[11];
  float* out = (float*)d_out;

  char* ws = (char*)d_ws;
  size_t off = 0;
  auto carve = [&](size_t bytes) -> char* {
    char* p = ws + off; off += (bytes + 255) & ~(size_t)255; return p;
  };
  bf16_t* xb  = (bf16_t*)carve((size_t)33554432 * 2);        // x in bf16 [B,H,W,C]
  bf16_t* qwB = (bf16_t*)carve((size_t)65536 * 2);           // [K=256][N=256]
  bf16_t* w0B = (bf16_t*)carve((size_t)16384 * 512 * 2);     // [K=16384][N=512]
  bf16_t* w1B = (bf16_t*)carve((size_t)4096 * 512 * 2);      // [K=4096][N=512]
  bf16_t* pwB = (bf16_t*)carve((size_t)512 * 256 * 2);       // [K=512][N=256]
  float*  qA  = (float*)carve((size_t)33554432 * 4);         // q f32 [B,H,W,C]
  float*  kv0 = (float*)carve((size_t)2048 * 512 * 4);       // level-0 KV (post-LN, in place)
  float*  kv1 = (float*)carve((size_t)8192 * 512 * 4);       // level-1 KV
  bf16_t* yb  = (bf16_t*)carve((size_t)131072 * 512 * 2);    // concat(y0,y1) bf16 [BHW][512]

  // pack / convert
  k_cvt_bf16<<<32768, 256, 0, stream>>>(x, xb, 8388608);
  k_pack_qw <<<256,   256, 0, stream>>>(qw, qwB);
  k_pack_conv<<<32768, 256, 0, stream>>>(cw0, w0B, 8);
  k_pack_conv<<<8192,  256, 0, stream>>>(cw1, w1B, 4);
  k_pack_pw <<<512,   256, 0, stream>>>(pw, pwB);

  // q = x @ qw^T : M=131072 N=256 K=256
  k_gemm<0><<<dim3(1024, 2), 256, 0, stream>>>(xb, qwB, qA, nullptr, 256, 256, 256, 256);

  // level-0 conv as patch-GEMM: M=2048 N=512 K=16384 (input = xb, pixel stride 256)
  k_gemm<1><<<dim3(16, 4), 256, 0, stream>>>(xb, w0B, kv0, nullptr, 16384, 256, 512, 512);
  k_bias_ln<<<256, 256, 0, stream>>>(kv0, cb0, g0, b0);
  k_attn<0><<<4096, 256, 0, stream>>>(qA, kv0, yb);

  // level-1 conv consumes level-0 attention output (yb channels [0,256), pixel stride 512)
  k_gemm<2><<<dim3(64, 4), 256, 0, stream>>>(yb, w1B, kv1, nullptr, 4096, 512, 512, 512);
  k_bias_ln<<<1024, 256, 0, stream>>>(kv1, cb1, g1, b1);
  k_attn<1><<<16384, 64, 0, stream>>>(qA, kv1, yb);

  // out = concat(y0,y1) @ pw^T + pb : M=131072 N=256 K=512
  k_gemm<0><<<dim3(1024, 2), 256, 0, stream>>>(yb, pwB, out, pb, 512, 512, 256, 256);
}
